// ConstrastiveIntraViewLucasLoss_89051851915600
// MI455X (gfx1250) — matrix operand order repaired and checked by reading, not compiled, not run on hardware
//
#include <hip/hip_runtime.h>
#include <math.h>

typedef __attribute__((ext_vector_type(16))) _Float16 v16h;
typedef __attribute__((ext_vector_type(8)))  float    v8f;

#define BATCH 8
#define CHN   768
#define HQ    128
#define WQ    128
#define IMG   (HQ*WQ)            // 16384 pixels per image
#define NPIX  (BATCH*IMG)        // 131072
#define NCLS  9
#define KCH   (CHN/32)           // 24 K-chunks of 32 channels
#define TEMPERATURE 0.1f
#define ALPHAV      0.99f
#define EPSV        1e-12f

// ---- workspace layout (bytes) ----
#define OFF_LABELS 0u                       // int32 [NPIX]            = 524288 B
#define OFF_COUNTS 524288u                  // float [16]
#define OFF_SUMS   524352u                  // float [9*768]           = 27648 B
#define OFF_RN     552000u                  // float [16]
#define OFF_PROTOA 552064u                  // f16   [24*32*16]        = 24576 B
#define OFF_S      576768u                  // float [9*NPIX]          = 4718592 B
// total ~5.30 MB

// ------------------------------------------------------------------
// K0: labels (downsample gt by 4) + per-class counts
// ------------------------------------------------------------------
__global__ void k_labels(const int* __restrict__ gt, int* __restrict__ labels,
                         float* __restrict__ counts) {
  __shared__ float lc[16];
  int t = threadIdx.x;
  if (t < 16) lc[t] = 0.f;
  __syncthreads();
  int n = blockIdx.x * 256 + t;
  int b = n >> 14, rem = n & (IMG - 1);
  int h = rem >> 7, w = rem & 127;
  int lab = gt[((b * 512 + h * 4) * 512) + w * 4];
  labels[n] = lab;
  atomicAdd(&lc[lab], 1.f);
  __syncthreads();
  if (t < 16 && lc[t] != 0.f) atomicAdd(&counts[t], lc[t]);
}

// ------------------------------------------------------------------
// K1: per-class channel sums.  Block = 256 threads, tile = 1024 pixels.
// LDS accumulator stride 772 (772 mod 64 == 4) -> the 9 classes land in
// 9 distinct banks for a given channel.
// ------------------------------------------------------------------
__global__ void k_segsum(const float* __restrict__ feats,
                         const int* __restrict__ labels,
                         float* __restrict__ sums) {
  __shared__ float acc[9 * 772];
  int t = threadIdx.x;
  for (int i = t; i < 9 * 772; i += 256) acc[i] = 0.f;
  __syncthreads();

  int pbase = blockIdx.x * 1024;            // tile never crosses an image (16384%1024==0)
  int b = pbase >> 14;
  int hw = pbase & (IMG - 1);

  int labb[4];
#pragma unroll
  for (int k = 0; k < 4; k++) labb[k] = labels[pbase + t + k * 256] * 772;

  const float* fp = feats + (size_t)b * CHN * IMG + hw + t;
  for (int c = 0; c < CHN; c++) {
    const float* fc = fp + (size_t)c * IMG;
    __builtin_prefetch(fc + 8 * (size_t)IMG, 0, 0);   // global_prefetch_b8, 8 channels ahead
#pragma unroll
    for (int k = 0; k < 4; k++) {
      float v = fc[k * 256];
      atomicAdd(&acc[labb[k] + c], v);                // ds_add_f32
    }
  }
  __syncthreads();
  for (int i = t; i < 9 * 768; i += 256) {
    int cls = i / 768, c = i - cls * 768;
    float v = acc[cls * 772 + c];
    if (v != 0.f) atomicAdd(&sums[i], v);
  }
}

// ------------------------------------------------------------------
// K2: EMA update + L2-normalize prototypes, then pack into the WMMA
// 16x32 f16 A-fragment lane layout: lane m=lane&15 holds row m;
// half i maps to K = i + 8*hi + (i>=8 ? 8 : 0), hi = lane>>4.
// ------------------------------------------------------------------
__global__ void k_protos(const float* __restrict__ proto_in,
                         const float* __restrict__ sums,
                         const float* __restrict__ counts,
                         _Float16* __restrict__ protoA) {
  __shared__ float PN[16 * 768];
  __shared__ float red[256];
  __shared__ float norms[16];
  int t = threadIdx.x;
  for (int i = t; i < 16 * 768; i += 256) PN[i] = 0.f;
  __syncthreads();

  for (int k = 0; k < NCLS; k++) {
    float cnt = counts[k];
    bool present = cnt > 0.f;
    float inv = 1.f / fmaxf(cnt, 1.f);
    float partial = 0.f;
    for (int c = t; c < 768; c += 256) {
      float p = proto_in[k * 768 + c];
      float u = present ? (ALPHAV * (sums[k * 768 + c] * inv) + (1.f - ALPHAV) * p) : p;
      PN[k * 768 + c] = u;
      partial += u * u;
    }
    red[t] = partial; __syncthreads();
    for (int s = 128; s > 0; s >>= 1) { if (t < s) red[t] += red[t + s]; __syncthreads(); }
    if (t == 0) norms[k] = fmaxf(sqrtf(red[0]), EPSV);
    __syncthreads();
  }
  for (int i = t; i < NCLS * 768; i += 256) { int k = i / 768; PN[i] *= (1.f / norms[k]); }
  __syncthreads();

  for (int idx = t; idx < KCH * 32 * 16; idx += 256) {
    int kc = idx >> 9;
    int lane = (idx >> 4) & 31;
    int i = idx & 15;
    int m = lane & 15, hi = lane >> 4;
    int ko = i + 8 * hi + ((i >= 8) ? 8 : 0);
    protoA[idx] = (_Float16)PN[m * 768 + kc * 32 + ko];
  }
}

// ------------------------------------------------------------------
// K3: s[9][NPIX] = protos @ feats^T via v_wmma_f32_16x16x32_f16.
// Each wave owns a 16-pixel column tile; B fragment loaded straight
// from global: per channel i, lanes 0-15 read 16 consecutive pixels
// (64B), lanes 16-31 read channel K+8 (another full 64B) -> features
// streamed exactly once, fully coalesced.
// ------------------------------------------------------------------
__global__ void k_scores(const float* __restrict__ feats,
                         const _Float16* __restrict__ protoA,
                         float* __restrict__ smat) {
  int wave = threadIdx.x >> 5;
  int lane = threadIdx.x & 31;
  int pixf = blockIdx.x * 128 + wave * 16;   // 8 waves * 16 pixels = 128/block
  int b = pixf >> 14;
  int hw = pixf & (IMG - 1);
  int col = lane & 15, hi = lane >> 4;

  const float* fbase = feats + (size_t)b * CHN * IMG + hw + col;
  const v16h* Ap = (const v16h*)protoA + lane;

  v8f acc = {};
  for (int kc = 0; kc < KCH; kc++) {
    v16h a = Ap[kc * 32];
    const float* fk = fbase + (size_t)(kc * 32 + hi * 8) * IMG;
    __builtin_prefetch(fk + 32 * (size_t)IMG, 0, 0);  // next K-chunk
    v16h bf;
#pragma unroll
    for (int i = 0; i < 16; i++) bf[i] = (_Float16)fk[(size_t)(i + ((i >= 8) ? 8 : 0)) * IMG];
    acc = __builtin_amdgcn_wmma_f32_16x16x32_f16(
        /*neg_a=*/false, a, /*neg_b=*/false, bf,
        /*c_mod=*/(short)0, acc, /*reuse_a=*/false, /*reuse_b=*/false);
  }
  // C/D layout: VGPR r -> row r (lanes 0-15) / row r+8 (lanes 16-31)
#pragma unroll
  for (int r = 0; r < 8; r++) {
    int m = r + hi * 8;
    if (m < NCLS) smat[(size_t)m * NPIX + pixf + col] = acc[r];
  }
}

// ------------------------------------------------------------------
// K4: rn2[m] = sum_n s[m][n]^2   (s lives in L2)
// ------------------------------------------------------------------
__global__ void k_rownorm(const float* __restrict__ smat, float* __restrict__ rn2) {
  __shared__ float red[256];
  int t = threadIdx.x;
  float part[NCLS];
#pragma unroll
  for (int m = 0; m < NCLS; m++) part[m] = 0.f;
  int base = blockIdx.x * 1024;
#pragma unroll
  for (int k = 0; k < 4; k++) {
    int p = base + t + k * 256;
#pragma unroll
    for (int m = 0; m < NCLS; m++) {
      float v = smat[(size_t)m * NPIX + p];
      part[m] += v * v;
    }
  }
  for (int m = 0; m < NCLS; m++) {
    red[t] = part[m]; __syncthreads();
    for (int s = 128; s > 0; s >>= 1) { if (t < s) red[t] += red[t + s]; __syncthreads(); }
    if (t == 0) atomicAdd(&rn2[m], red[0]);
    __syncthreads();
  }
}

// ------------------------------------------------------------------
// K5: log-softmax over classes, pick at label, mean -> loss
// ------------------------------------------------------------------
__global__ void k_loss(const float* __restrict__ smat, const int* __restrict__ labels,
                       const float* __restrict__ rn2, float* __restrict__ out) {
  __shared__ float inv[NCLS];
  __shared__ float red[256];
  int t = threadIdx.x;
  if (t < NCLS) inv[t] = 1.f / (fmaxf(sqrtf(rn2[t]), EPSV) * TEMPERATURE);
  __syncthreads();
  int n = blockIdx.x * 256 + t;
  int lab = labels[n];
  float l[NCLS], mx = -3.0e38f;
#pragma unroll
  for (int m = 0; m < NCLS; m++) {
    l[m] = smat[(size_t)m * NPIX + n] * inv[m];
    mx = fmaxf(mx, l[m]);
  }
  float se = 0.f, pick = 0.f;
#pragma unroll
  for (int m = 0; m < NCLS; m++) {
    se += expf(l[m] - mx);
    if (m == lab) pick = l[m];
  }
  float picked = (pick - mx) - logf(se);
  red[t] = picked; __syncthreads();
  for (int s = 128; s > 0; s >>= 1) { if (t < s) red[t] += red[t + s]; __syncthreads(); }
  if (t == 0) atomicAdd(out, -red[0] * (1.f / (float)NPIX));
}

// ------------------------------------------------------------------
extern "C" void kernel_launch(void* const* d_in, const int* in_sizes, int n_in,
                              void* d_out, int out_size, void* d_ws, size_t ws_size,
                              hipStream_t stream) {
  const float* feats = (const float*)d_in[0];
  const int*   gt    = (const int*)d_in[1];
  const float* proto = (const float*)d_in[2];
  char* ws = (char*)d_ws;

  int*      labels = (int*)(ws + OFF_LABELS);
  float*    counts = (float*)(ws + OFF_COUNTS);
  float*    sums   = (float*)(ws + OFF_SUMS);
  float*    rn2    = (float*)(ws + OFF_RN);
  _Float16* protoA = (_Float16*)(ws + OFF_PROTOA);
  float*    smat   = (float*)(ws + OFF_S);
  float*    out    = (float*)d_out;

  // zero accumulators (counts + sums + rownorm) and the scalar output
  hipMemsetAsync(ws + OFF_COUNTS, 0, OFF_PROTOA - OFF_COUNTS, stream);
  hipMemsetAsync(d_out, 0, sizeof(float), stream);

  k_labels <<<NPIX / 256,  256, 0, stream>>>(gt, labels, counts);
  k_segsum <<<NPIX / 1024, 256, 0, stream>>>(feats, labels, sums);
  k_protos <<<1,           256, 0, stream>>>(proto, sums, counts, protoA);
  k_scores <<<NPIX / 128,  256, 0, stream>>>(feats, protoA, smat);
  k_rownorm<<<128,         256, 0, stream>>>(smat, rn2);
  k_loss   <<<NPIX / 256,  256, 0, stream>>>(smat, labels, rn2, out);
}
